// Attention_45569603010584
// MI455X (gfx1250) — matrix operand order repaired and checked by reading, not compiled, not run on hardware
//
#include <hip/hip_runtime.h>

// Problem constants: B=32, C=64, N=207, T=96, H=32, K=3
#define BB 32
#define CC 64
#define NDIM 207
#define TLEN 96
#define HH 32

typedef __attribute__((ext_vector_type(16))) _Float16 v16h;
typedef __attribute__((ext_vector_type(8)))  float    v8f;

// ---------------- LDS plan (bytes), phase-aliased, peak 63488 ----------------
// [S]=stage, [A]=qkv, [BC]=attn+softmax (+weight cvt), [D]=AO, [E]=proj+res,
// [F]=ff1, [G]=ff2
#define OFF_QT    0      // f16 qT[96][32] row-major          [A..BC]
#define OFF_AOSW  0      // f16 AO  B-swz [6][32][16]         [D..E]
#define OFF_KSW   6144   // f16 k   B-swz [6][32][16]         [A..BC]
#define OFF_H1SW  6144   // f16 h1  B-swz [2][6][32][16]      [F..G]
#define OFF_VSW   12288  // f16 v   B-swz [3][2][32][16]      [A..D]
#define OFF_XSW   18432  // f16 x   B-swz [2][6][32][16]      [S..A]
#define OFF_PSW   18432  // f16 P   A-swz [6][3][32][16]      [BC..D]
#define OFF_OFSW  18432  // f32 of  B-swz [2][6][32][16]      [E..G]
#define OFF_QWH   30720  // f16 qw  A-swz [3][2][2][32][16]   [S..A]
#define OFF_KWH   43008  // f16 kw  A-swz [3][2][2][32][16]   [S..A]
#define OFF_W1H   43008  // f16 W1  A-swz [4][2][32][16]      [BC..F]
#define OFF_W2H   51200  // f16 W2  A-swz [4][2][32][16]      [BC..G]
#define OFF_VWH   55296  // f16 vw  A-swz [2][2][32][16]      [S..A]
#define OFF_OWH   59392  // f16 ow  A-swz [4][1][32][16]      [BC..E]
#define SMEM_BYTES 63488

static __device__ __forceinline__ v8f wmma16(v16h a, v16h b, v8f c) {
  return __builtin_amdgcn_wmma_f32_16x16x32_f16(false, a, false, b, (short)0, c, false, false);
}

// Swizzled tile (512 halves = 32 lanes x 16) -> single aligned 32B v16h load
static __device__ __forceinline__ v16h ldTile(const _Float16* p, int tile, int lane) {
  return *(const v16h*)(p + (tile << 9) + (lane << 4));
}
// f32 B-swz tile, convert to f16 on load (4x b128 + packed cvt)
static __device__ __forceinline__ v16h ldTileF32(const float* p, int tile, int lane) {
  const float* s = p + (tile << 9) + (lane << 4);
  v16h b;
#pragma unroll
  for (int e = 0; e < 16; ++e) b[e] = (_Float16)s[e];
  return b;
}
// x B-tile with causal temporal shift (zero pad t<0); lane keeps its k-half role
static __device__ __forceinline__ v16h ldXshift(const _Float16* xsw, int kt, int nt, int shift, int lane) {
  const int t = nt * 16 + (lane & 15) + shift;
  v16h b = {};
  if (t >= 0) {
    const int l2 = (t & 15) + ((lane >> 4) << 4);
    b = *(const v16h*)(xsw + (((kt * 6 + (t >> 4)) * 32 + l2) << 4));
  }
  return b;
}
// A gather from row-major f16 (k run of 16 contiguous -> 32B ds load)
static __device__ __forceinline__ v16h ldA_rm(const _Float16* s, int ld, int m0, int lane) {
  const int m = m0 + (lane & 15), hi = lane >> 4;
  v16h a;
#pragma unroll
  for (int e = 0; e < 16; ++e)
    a[e] = s[m * ld + (hi << 4) + e];
  return a;
}

__global__ __launch_bounds__(256)
void attn_tcn_kernel(const float* __restrict__ xg, const float* __restrict__ qw,
                     const float* __restrict__ kw, const float* __restrict__ vw,
                     const float* __restrict__ ow, const float* __restrict__ w1,
                     const float* __restrict__ b1, const float* __restrict__ w2,
                     const float* __restrict__ b2, float* __restrict__ outg) {
  __shared__ __align__(64) char smem[SMEM_BYTES];
  _Float16* QT   = (_Float16*)(smem + OFF_QT);
  _Float16* AOsw = (_Float16*)(smem + OFF_AOSW);
  _Float16* Ksw  = (_Float16*)(smem + OFF_KSW);
  _Float16* H1sw = (_Float16*)(smem + OFF_H1SW);
  _Float16* Vsw  = (_Float16*)(smem + OFF_VSW);
  _Float16* Xsw  = (_Float16*)(smem + OFF_XSW);
  _Float16* Psw  = (_Float16*)(smem + OFF_PSW);
  float*    OFsw = (float*)   (smem + OFF_OFSW);
  _Float16* QWh  = (_Float16*)(smem + OFF_QWH);
  _Float16* KWh  = (_Float16*)(smem + OFF_KWH);
  _Float16* W1h  = (_Float16*)(smem + OFF_W1H);
  _Float16* W2h  = (_Float16*)(smem + OFF_W2H);
  _Float16* VWh  = (_Float16*)(smem + OFF_VWH);
  _Float16* OWh  = (_Float16*)(smem + OFF_OWH);

  const int tid  = threadIdx.x;
  const int lane = tid & 31;
  const int wave = tid >> 5;
  const int bn   = blockIdx.x;
  const int bIdx = bn / NDIM;
  const int node = bn % NDIM;
  const size_t base = (size_t)bIdx * CC * NDIM * TLEN + (size_t)node * TLEN;
  const int cstride = NDIM * TLEN;

  // ---------------- Phase S: stage x + conv weights (swizzled f16) ----------
  for (int i = tid; i < CC * TLEN; i += 256) {          // x -> B-swz
    const int c = i / TLEN, t = i % TLEN;
    const float v = xg[base + (size_t)c * cstride + t];
    const int l = (t & 15) + (((c >> 4) & 1) << 4);
    Xsw[((((c >> 5) * 6 + (t >> 4)) * 32 + l) << 4) + (c & 15)] = (_Float16)v;
  }
  for (int i = tid; i < 6144; i += 256) {               // q_w, k_w -> A-swz per tap
    const int tl = i >> 9, l = (i >> 4) & 31, e = i & 15;
    const int kt = tl & 1, mt = (tl >> 1) & 1, d = tl >> 2;
    const int m = mt * 16 + (l & 15);
    const int k = kt * 32 + ((l >> 4) << 4) + e;
    QWh[i] = (_Float16)qw[(m * CC + k) * 3 + d];
    KWh[i] = (_Float16)kw[(m * CC + k) * 3 + d];
  }
  for (int i = tid; i < 2048; i += 256) {               // v_w -> A-swz
    const int tl = i >> 9, l = (i >> 4) & 31, e = i & 15;
    const int kt = tl & 1, mt = tl >> 1;
    const int m = mt * 16 + (l & 15);
    const int k = kt * 32 + ((l >> 4) << 4) + e;
    VWh[i] = (_Float16)vw[m * CC + k];
  }
  __syncthreads();

  // ---------------- Phase A: q, k (3-tap causal conv), v -------------------
  for (int j = wave; j < 36; j += 8) {
    const int kind = j / 12, rr = j % 12, mt = rr / 6, nt = rr % 6;
    const int n = lane & 15, dhi = lane >> 4;
    v8f acc = {};
    if (kind < 2) {
      const _Float16* wsw = kind ? KWh : QWh;
#pragma unroll
      for (int d = 0; d < 3; ++d)
#pragma unroll
        for (int kt = 0; kt < 2; ++kt) {
          v16h a = ldTile(wsw, (d * 2 + mt) * 2 + kt, lane);
          v16h b = ldXshift(Xsw, kt, nt, d - 2, lane);
          acc = wmma16(a, b, acc);
        }
      if (kind == 0) {                                  // qT[t][h] row-major (A for BC)
#pragma unroll
        for (int r = 0; r < 8; ++r)
          QT[(nt * 16 + n) * HH + mt * 16 + r + 8 * dhi] = (_Float16)acc[r];
      } else {                                          // k -> B-swz (k-dim = h)
#pragma unroll
        for (int r = 0; r < 8; ++r) {
          const int h = mt * 16 + r + 8 * dhi, t = nt * 16 + n;
          Ksw[((((t >> 4) * 32) + (t & 15) + ((h >> 4) << 4)) << 4) + (h & 15)] = (_Float16)acc[r];
        }
      }
    } else {
#pragma unroll
      for (int kt = 0; kt < 2; ++kt) {
        v16h a = ldTile(VWh, mt * 2 + kt, lane);
        v16h b = ldTile(Xsw, kt * 6 + nt, lane);
        acc = wmma16(a, b, acc);
      }
#pragma unroll
      for (int r = 0; r < 8; ++r) {                     // v -> B-swz (k-dim = t, n = h)
        const int h = mt * 16 + r + 8 * dhi, t = nt * 16 + n;
        const int l = (h & 15) + (((t >> 4) & 1) << 4);
        Vsw[(((((t >> 5) * 2) + (h >> 4)) * 32 + l) << 4) + (t & 15)] = (_Float16)acc[r];
      }
    }
  }
  __syncthreads();

  // ------- Phase BC: causal attn logits + fused softmax (waves 0..5) -------
  //         waves 6..7: convert o_w / W1 / W2 -> A-swz f16
  if (wave < 6) {
    const int tq = wave;
    const int n = lane & 15, dhi = lane >> 4;
    v16h aQ = ldA_rm(QT, HH, tq * 16, lane);            // A[m=t][k=h]
    v8f acc[6];
#pragma unroll
    for (int tk = 0; tk < 6; ++tk) acc[tk] = (v8f){};
#pragma unroll
    for (int tk = 0; tk < 6; ++tk)
      if (tk <= tq) acc[tk] = wmma16(aQ, ldTile(Ksw, tk, lane), acc[tk]);
    // mask diagonal tile + per-lane row max
    float mx[8], sm[8];
#pragma unroll
    for (int r = 0; r < 8; ++r) mx[r] = -3.0e38f;
#pragma unroll
    for (int tk = 0; tk < 6; ++tk)
      if (tk <= tq)
#pragma unroll
        for (int r = 0; r < 8; ++r) {
          float s = acc[tk][r];
          if (tk == tq && n > r + 8 * dhi) s = -3.0e38f;
          acc[tk][r] = s;
          mx[r] = fmaxf(mx[r], s);
        }
#pragma unroll
    for (int r = 0; r < 8; ++r)
#pragma unroll
      for (int off = 1; off < 16; off <<= 1)
        mx[r] = fmaxf(mx[r], __shfl_xor(mx[r], off, 32));
#pragma unroll
    for (int r = 0; r < 8; ++r) sm[r] = 0.f;
#pragma unroll
    for (int tk = 0; tk < 6; ++tk)
      if (tk <= tq)
#pragma unroll
        for (int r = 0; r < 8; ++r) sm[r] += __expf(acc[tk][r] - mx[r]);
#pragma unroll
    for (int r = 0; r < 8; ++r) {
#pragma unroll
      for (int off = 1; off < 16; off <<= 1) sm[r] += __shfl_xor(sm[r], off, 32);
      sm[r] = 1.f / sm[r];
    }
    // scatter probabilities -> A-swz P (plus zero-fill odd half of diag k-tile)
#pragma unroll
    for (int tk = 0; tk < 6; ++tk)
      if (tk <= tq)
#pragma unroll
        for (int r = 0; r < 8; ++r) {
          const float p = __expf(acc[tk][r] - mx[r]) * sm[r];
          const int kg = tk * 16 + n, koff = kg & 31;
          const int e = koff & 15;
          const int l = (r + 8 * dhi) + ((koff >> 4) << 4);
          Psw[(((tq * 3 + (kg >> 5)) * 32 + l) << 4) + e] = (_Float16)p;
        }
    if ((tq & 1) == 0) {
      const int kg = (tq + 1) * 16 + n, koff = kg & 31;
      const int e = koff & 15;
#pragma unroll
      for (int r = 0; r < 8; ++r)
        Psw[(((tq * 3 + (kg >> 5)) * 32 + (r + 8 * dhi) + ((koff >> 4) << 4)) << 4) + e] =
            (_Float16)0;
    }
  } else {
    const int st = tid - 192;                           // 64 threads
    for (int i = st; i < 2048; i += 64) {               // o_w [64][32] -> A-swz
      const int tl = i >> 9, l = (i >> 4) & 31, e = i & 15;
      const int m = tl * 16 + (l & 15);
      const int k = ((l >> 4) << 4) + e;
      OWh[i] = (_Float16)ow[m * HH + k];
    }
    for (int i = st; i < 4096; i += 64) {               // W1, W2 [64][64] -> A-swz
      const int tl = i >> 9, l = (i >> 4) & 31, e = i & 15;
      const int kt = tl & 1, mt = tl >> 1;
      const int m = mt * 16 + (l & 15);
      const int k = kt * 32 + ((l >> 4) << 4) + e;
      W1h[i] = (_Float16)w1[m * CC + k];
      W2h[i] = (_Float16)w2[m * CC + k];
    }
  }
  __syncthreads();

  // ---------------- Phase D: AO = P x v (skip zero causal tiles) -----------
  for (int j = wave; j < 12; j += 8) {
    const int tq = j >> 1, ht = j & 1;
    const int n = lane & 15, dhi = lane >> 4;
    v8f acc = {};
#pragma unroll
    for (int kt = 0; kt < 3; ++kt)
      if (kt <= (tq >> 1))
        acc = wmma16(ldTile(Psw, tq * 3 + kt, lane), ldTile(Vsw, kt * 2 + ht, lane), acc);
#pragma unroll
    for (int r = 0; r < 8; ++r) {                       // AO -> B-swz (k-dim = h, n = t)
      const int t = tq * 16 + r + 8 * dhi, h = ht * 16 + n;
      AOsw[((((t >> 4) * 32) + (t & 15) + ((h >> 4) << 4)) << 4) + (h & 15)] = (_Float16)acc[r];
    }
  }
  __syncthreads();

  // ---------------- Phase E: out_f = x + o_w x AO  (f32 B-swz) -------------
  for (int j = wave; j < 24; j += 8) {
    const int ct = j / 6, tt = j % 6;
    const int n = lane & 15, dhi = lane >> 4;
    v8f acc = {};
    acc = wmma16(ldTile(OWh, ct, lane), ldTile(AOsw, tt, lane), acc);
    const int t = tt * 16 + n;
#pragma unroll
    for (int r = 0; r < 8; ++r) {
      const int c = ct * 16 + r + 8 * dhi;
      const float val = acc[r] + xg[base + (size_t)c * cstride + t];
      const int l = (t & 15) + (((c >> 4) & 1) << 4);
      OFsw[((((c >> 5) * 6 + tt) * 32 + l) << 4) + (c & 15)] = val;
    }
  }
  __syncthreads();

  // ---------------- Phase F: h1 = relu(W1 x out_f + b1) --------------------
  for (int j = wave; j < 24; j += 8) {
    const int dt = j / 6, tt = j % 6;
    const int n = lane & 15, dhi = lane >> 4;
    v8f acc = {};
#pragma unroll
    for (int kt = 0; kt < 2; ++kt)
      acc = wmma16(ldTile(W1h, dt * 2 + kt, lane), ldTileF32(OFsw, kt * 6 + tt, lane), acc);
    const int t = tt * 16 + n;
#pragma unroll
    for (int r = 0; r < 8; ++r) {
      const int d = dt * 16 + r + 8 * dhi;
      const float h = fmaxf(acc[r] + b1[d], 0.f);
      const int l = (t & 15) + (((d >> 4) & 1) << 4);
      H1sw[((((d >> 5) * 6 + tt) * 32 + l) << 4) + (d & 15)] = (_Float16)h;
    }
  }
  __syncthreads();

  // ---------------- Phase G: out = out_f + W2 x h1 + b2 --------------------
  for (int j = wave; j < 24; j += 8) {
    const int ct = j / 6, tt = j % 6;
    const int n = lane & 15, dhi = lane >> 4;
    v8f acc = {};
#pragma unroll
    for (int kt = 0; kt < 2; ++kt)
      acc = wmma16(ldTile(W2h, ct * 2 + kt, lane), ldTile(H1sw, kt * 6 + tt, lane), acc);
    const int t = tt * 16 + n;
#pragma unroll
    for (int r = 0; r < 8; ++r) {
      const int c = ct * 16 + r + 8 * dhi;
      const int l = (t & 15) + (((c >> 4) & 1) << 4);
      const float of = OFsw[((((c >> 5) * 6 + tt) * 32 + l) << 4) + (c & 15)];
      outg[base + (size_t)c * cstride + t] = acc[r] + b2[c] + of;
    }
  }
}

extern "C" void kernel_launch(void* const* d_in, const int* in_sizes, int n_in,
                              void* d_out, int out_size, void* d_ws, size_t ws_size,
                              hipStream_t stream) {
  (void)in_sizes; (void)n_in; (void)d_ws; (void)ws_size; (void)out_size;
  const float* x  = (const float*)d_in[0];
  const float* qw = (const float*)d_in[1];
  const float* kw = (const float*)d_in[2];
  const float* vw = (const float*)d_in[3];
  const float* ow = (const float*)d_in[4];
  const float* w1 = (const float*)d_in[5];
  const float* b1 = (const float*)d_in[6];
  const float* w2 = (const float*)d_in[7];
  const float* b2 = (const float*)d_in[8];
  float* out = (float*)d_out;

  dim3 grid(BB * NDIM);   // 6624 (b,n) items, one [64x96] slice each
  dim3 block(256);        // 8 wave32 waves
  attn_tcn_kernel<<<grid, block, 0, stream>>>(x, qw, kw, vw, ow, w1, b1, w2, b2, out);
}
